// LlamaMHALayer_43456479101527
// MI455X (gfx1250) — compile-verified, matrix operand order checked by reading, not run on hardware
//
#include <hip/hip_runtime.h>
#include <cstdint>

// ---------------- model constants ----------------
constexpr int DM   = 2048;   // d_model
constexpr int NH   = 32;     // heads
constexpr int NKV  = 8;      // kv heads
constexpr int HD   = 64;     // head dim
constexpr int DFF  = 8192;
constexpr int SEQ  = 2048;
constexpr int BB   = 2;
constexpr int MTOK = BB * SEQ;            // 4096 rows
constexpr int QKVN = (NH + 2 * NKV) * HD; // 3072

// ---------------- types ----------------
typedef __attribute__((ext_vector_type(16))) __bf16        bf16x16;
typedef __attribute__((ext_vector_type(8)))  float         f32x8;
typedef __attribute__((ext_vector_type(4)))  unsigned int  u32x4;
typedef __attribute__((ext_vector_type(8)))  int           i32x8;
typedef __attribute__((ext_vector_type(4)))  int           i32x4;

#if defined(__has_builtin)
#if __has_builtin(__builtin_amdgcn_tensor_load_to_lds) && \
    __has_builtin(__builtin_amdgcn_s_wait_tensorcnt)
#define HAVE_TDM 1
#endif
#endif
#ifndef HAVE_TDM
#define HAVE_TDM 0
#endif

__device__ inline unsigned short f2bf(float f) {
  unsigned int u = __float_as_uint(f);
  unsigned int r = u + 0x7FFFu + ((u >> 16) & 1u);   // round-to-nearest-even
  return (unsigned short)(r >> 16);
}

__device__ inline f32x8 wmma_bf16(bf16x16 a, bf16x16 b, f32x8 c) {
  // D = A(16x32 bf16) * B(32x16 bf16) + C(16x16 f32)
  return __builtin_amdgcn_wmma_f32_16x16x32_bf16(false, a, false, b,
                                                 (short)0, c, false, false);
}

// ---------------- weight fp32 [K,N] -> bf16 [N,K] ----------------
__global__ __launch_bounds__(256)
void transpose_cast(const float* __restrict__ in, unsigned short* __restrict__ out,
                    int K, int N) {
  __shared__ float tile[32][33];
  const int n0 = blockIdx.x * 32, k0 = blockIdx.y * 32;
  const int tx = threadIdx.x, ty = threadIdx.y;   // (32, 8)
  for (int i = ty; i < 32; i += 8)
    tile[i][tx] = in[(size_t)(k0 + i) * N + n0 + tx];
  __syncthreads();
  for (int i = ty; i < 32; i += 8)
    out[(size_t)(n0 + i) * K + k0 + tx] = f2bf(tile[tx][i]);
}

// ---------------- RMSNorm fp32 -> bf16 ----------------
__global__ __launch_bounds__(256)
void rmsnorm_bf16(const float* __restrict__ X, const float* __restrict__ W,
                  unsigned short* __restrict__ H) {
  const int row = blockIdx.x;
  const int tid = threadIdx.x;
  const float* x = X + (size_t)row * DM;
  float v[8];
  float p = 0.f;
#pragma unroll
  for (int i = 0; i < 8; ++i) { v[i] = x[tid + i * 256]; p += v[i] * v[i]; }
#pragma unroll
  for (int off = 16; off; off >>= 1) p += __shfl_xor(p, off, 32);
  __shared__ float red[8];
  if ((tid & 31) == 0) red[tid >> 5] = p;
  __syncthreads();
  float tot = 0.f;
#pragma unroll
  for (int i = 0; i < 8; ++i) tot += red[i];
  const float rn = rsqrtf(tot / (float)DM + 1e-5f);
#pragma unroll
  for (int i = 0; i < 8; ++i)
    H[(size_t)row * DM + tid + i * 256] = f2bf(v[i] * rn * W[tid + i * 256]);
}

// ---------------- TDM tile load: 128 rows x 32 bf16, row-padded in LDS -------
// D# per CDNA5 ISA 8.3/8.4: 2D tile, data_size=2B, pad 16B after each 64B row
// -> LDS row stride 80B (LDT=40 ushorts), matching the WMMA fragment reads.
#if HAVE_TDM
__device__ inline void tdm_load_tile_128x32(const unsigned short* gptr,
                                            unsigned int lds_byte_off, int K) {
  const unsigned long long ga = (unsigned long long)(uintptr_t)gptr;
  u32x4 g0;
  g0[0] = 1u;                                        // count=1 (valid), load
  g0[1] = lds_byte_off;                              // lds_addr
  g0[2] = (unsigned int)(ga & 0xFFFFFFFFu);          // global_addr[31:0]
  g0[3] = (unsigned int)((ga >> 32) & 0x01FFFFFFu)   // global_addr[56:32]
          | 0x80000000u;                             // type=2 ("image")
  i32x8 g1;
  g1[0] = (1 << 16)      // data_size = 2 bytes
        | (1 << 20)      // pad_enable
        | (3 << 22)      // pad_interval: 16 DWORDs (= 64B row)
        | (3 << 25);     // pad_amount:   4 DWORDs (= 16B pad)
  g1[1] = (K & 0xFFFF) << 16;                        // tensor_dim0[15:0]
  g1[2] = ((K >> 16) & 0xFFFF) | (128 << 16);        // tensor_dim0[31:16], tensor_dim1 lo
  g1[3] = (32 << 16);                                // tensor_dim1 hi=0, tile_dim0=32
  g1[4] = 128;                                       // tile_dim1=128, tile_dim2=0
  g1[5] = K;                                         // tensor_dim0_stride[31:0]
  g1[6] = 0;                                         // stride hi, dim1_stride lo
  g1[7] = 0;
  const i32x4 zz4 = {0, 0, 0, 0};                    // groups 2/3 unused (2D tile)
  const i32x8 zz8 = {0, 0, 0, 0, 0, 0, 0, 0};
  __builtin_amdgcn_tensor_load_to_lds(g0, g1, zz4, zz4, zz8, 0);
}
#endif

// ---------------- WMMA GEMM: C[M,N] = A[M,K](bf16) * Bt[N,K](bf16)^T (+res) ----
// block tile 128x128, 8 waves (4x2), wave tile 32x64, K-step 32.
// TDM path: double-buffered tiles fetched by the Tensor Data Mover.
constexpr int LDT      = 40;         // padded ushort row stride (80B)
constexpr int TILE_USH = 128 * LDT;  // 5120 ushorts = 10240 B per tile
constexpr unsigned GEMM_LDS_BYTES = 4 * TILE_USH * 2;  // A0,B0,A1,B1

template <bool ADD_RES>
__global__ __launch_bounds__(256)
void gemm_bf16_wmma(const unsigned short* __restrict__ A,
                    const unsigned short* __restrict__ Bt,
                    const float* __restrict__ res,
                    float* __restrict__ C,
                    int M, int N, int K) {
  extern __shared__ __align__(16) unsigned short smem[];

  const int tid  = threadIdx.x;
  const int lane = tid & 31;
  const int wave = tid >> 5;
  const int wm   = (wave >> 1) * 32;            // 0,32,64,96
  const int wn   = (wave & 1) * 64;             // 0,64
  const int m0   = blockIdx.y * 128;
  const int n0   = blockIdx.x * 128;
  const int ln   = lane & 15;
  const int hi   = lane >> 4;

  f32x8 acc[2][4];
  const f32x8 z = {0.f, 0.f, 0.f, 0.f, 0.f, 0.f, 0.f, 0.f};
#pragma unroll
  for (int i = 0; i < 2; ++i)
#pragma unroll
    for (int j = 0; j < 4; ++j) acc[i][j] = z;

#if HAVE_TDM
  // prologue: fetch first K-tile into buffer 0 (one TDM issue per tile, wave 0)
  if (wave == 0) {
    tdm_load_tile_128x32(A  + (size_t)m0 * K, 0u, K);
    tdm_load_tile_128x32(Bt + (size_t)n0 * K, (unsigned)TILE_USH * 2u, K);
  }
  int buf = 0;
  for (int kk = 0; kk < K; kk += 32) {
    if (wave == 0) __builtin_amdgcn_s_wait_tensorcnt(0);
    __syncthreads();                      // current buf ready; prev compute done
    if (wave == 0 && kk + 32 < K) {
      const unsigned nb = (unsigned)(buf ^ 1) * (unsigned)(2 * TILE_USH) * 2u;
      tdm_load_tile_128x32(A  + (size_t)m0 * K + kk + 32, nb, K);
      tdm_load_tile_128x32(Bt + (size_t)n0 * K + kk + 32, nb + TILE_USH * 2u, K);
    }
    const unsigned short* pA = smem + buf * 2 * TILE_USH;
    const unsigned short* pB = pA + TILE_USH;

    bf16x16 af[2], bfv[4];
#pragma unroll
    for (int i = 0; i < 2; ++i)
      af[i] = *(const bf16x16*)(pA + (wm + i * 16 + ln) * LDT + hi * 16);
#pragma unroll
    for (int j = 0; j < 4; ++j)
      bfv[j] = *(const bf16x16*)(pB + (wn + j * 16 + ln) * LDT + hi * 16);
#pragma unroll
    for (int i = 0; i < 2; ++i)
#pragma unroll
      for (int j = 0; j < 4; ++j)
        acc[i][j] = wmma_bf16(af[i], bfv[j], acc[i][j]);
    buf ^= 1;
  }
#else
  // fallback: per-thread cooperative copies
  unsigned short* As = smem;
  unsigned short* Bs = smem + TILE_USH;
  const int lrow = tid >> 1;                    // 0..127
  const int lcol = (tid & 1) * 16;              // 0 / 16
  for (int kk = 0; kk < K; kk += 32) {
    const uint4* ga = (const uint4*)(A  + (size_t)(m0 + lrow) * K + kk + lcol);
    const uint4* gb = (const uint4*)(Bt + (size_t)(n0 + lrow) * K + kk + lcol);
    uint4* sa = (uint4*)(As + lrow * LDT + lcol);
    uint4* sb = (uint4*)(Bs + lrow * LDT + lcol);
    sa[0] = ga[0]; sa[1] = ga[1];
    sb[0] = gb[0]; sb[1] = gb[1];
    __syncthreads();
    bf16x16 af[2], bfv[4];
#pragma unroll
    for (int i = 0; i < 2; ++i)
      af[i] = *(const bf16x16*)(As + (wm + i * 16 + ln) * LDT + hi * 16);
#pragma unroll
    for (int j = 0; j < 4; ++j)
      bfv[j] = *(const bf16x16*)(Bs + (wn + j * 16 + ln) * LDT + hi * 16);
#pragma unroll
    for (int i = 0; i < 2; ++i)
#pragma unroll
      for (int j = 0; j < 4; ++j)
        acc[i][j] = wmma_bf16(af[i], bfv[j], acc[i][j]);
    __syncthreads();
  }
#endif

  // C/D layout: lanes 0-15: M=r, lanes 16-31: M=r+8; N = lane&15
#pragma unroll
  for (int i = 0; i < 2; ++i)
#pragma unroll
    for (int j = 0; j < 4; ++j)
#pragma unroll
      for (int r = 0; r < 8; ++r) {
        const size_t row = (size_t)(m0 + wm + i * 16 + hi * 8 + r);
        const size_t col = (size_t)(n0 + wn + j * 16 + ln);
        float v = acc[i][j][r];
        if (ADD_RES) v += res[row * N + col];
        C[row * N + col] = v;
      }
}

// ---------------- RoPE + layout: qkv fp32 -> Q[B,H,S,64], K[B,KV,S,64], V^T[B,KV,64,S]
__device__ inline float rope_val(const float* xp, int d, int s) {
  const int j = d & 31;
  const float inv = __expf(-(float)(2 * j) * (9.210340371976184f / 64.0f));
  const float ang = (float)s * inv;
  float sn, c;
  __sincosf(ang, &sn, &c);
  const float x1 = xp[j], x2 = xp[j + 32];
  return (d < 32) ? (x1 * c - x2 * sn) : (x1 * sn + x2 * c);
}

__global__ __launch_bounds__(64)
void rope_layout(const float* __restrict__ qkv,
                 unsigned short* __restrict__ Qo,
                 unsigned short* __restrict__ Ko,
                 unsigned short* __restrict__ Vt) {
  const int blk = blockIdx.x;          // b*SEQ*48 + s*48 + hh
  const int d   = threadIdx.x;         // 0..63
  const int hh  = blk % 48;
  const int s   = (blk / 48) % SEQ;
  const int b   = blk / (48 * SEQ);
  const float* row = qkv + (size_t)(b * SEQ + s) * QKVN;
  if (hh < NH) {
    const float* xp = row + hh * HD;
    Qo[((size_t)(b * NH + hh) * SEQ + s) * HD + d] = f2bf(rope_val(xp, d, s));
  } else if (hh < NH + NKV) {
    const int kh = hh - NH;
    const float* xp = row + NH * HD + kh * HD;
    Ko[((size_t)(b * NKV + kh) * SEQ + s) * HD + d] = f2bf(rope_val(xp, d, s));
  } else {
    const int vh = hh - NH - NKV;
    const float* xp = row + (NH + NKV) * HD + vh * HD;
    Vt[((size_t)(b * NKV + vh) * HD + d) * SEQ + s] = f2bf(xp[d]);
  }
}

// ---------------- flash attention: one wave per 16-query block ----------------
__global__ __launch_bounds__(128)
void attn_wmma(const unsigned short* __restrict__ Q,   // [B][H][S][64]
               const unsigned short* __restrict__ Kb,  // [B][KV][S][64]
               const unsigned short* __restrict__ Vt,  // [B][KV][64][S]
               unsigned short* __restrict__ ctx) {     // [B][S][H*64]
  __shared__ __align__(16) unsigned short Pt[4][16 * 32];
  const int tid  = threadIdx.x;
  const int lane = tid & 31;
  const int wv   = tid >> 5;
  const int wid  = blockIdx.x * 4 + wv;
  constexpr int QB = SEQ / 16;
  const int qblk = wid % QB;
  const int h    = (wid / QB) % NH;
  const int b    = wid / (QB * NH);
  const int kh   = h / (NH / NKV);
  const int q0   = qblk * 16;
  const int ln   = lane & 15;
  const int hi   = lane >> 4;

  const unsigned short* qptr =
      Q + ((size_t)(b * NH + h) * SEQ + q0 + ln) * HD + hi * 16;
  const bf16x16 qa0 = *(const bf16x16*)(qptr);        // dims 0..31
  const bf16x16 qa1 = *(const bf16x16*)(qptr + 32);   // dims 32..63

  const unsigned short* kbase = Kb + (size_t)(b * NKV + kh) * SEQ * HD;
  const unsigned short* vbase = Vt + (size_t)(b * NKV + kh) * HD * SEQ;
  unsigned short* pt = Pt[wv];

  const f32x8 z = {0.f, 0.f, 0.f, 0.f, 0.f, 0.f, 0.f, 0.f};
  f32x8 o[4] = {z, z, z, z};
  float mrow[8], lrow[8];
#pragma unroll
  for (int r = 0; r < 8; ++r) { mrow[r] = -1e30f; lrow[r] = 0.f; }

  for (int kb = 0; kb <= q0 + 15; kb += 32) {
    f32x8 s[2] = {z, z};
#pragma unroll
    for (int f = 0; f < 2; ++f) {
      const unsigned short* kp = kbase + (size_t)(kb + f * 16 + ln) * HD + hi * 16;
      const bf16x16 kf0 = *(const bf16x16*)(kp);        // dims 0..31
      const bf16x16 kf1 = *(const bf16x16*)(kp + 32);   // dims 32..63
      s[f] = wmma_bf16(qa0, kf0, s[f]);
      s[f] = wmma_bf16(qa1, kf1, s[f]);
    }
    const int qbase = q0 + hi * 8;
#pragma unroll
    for (int f = 0; f < 2; ++f) {
      const int key = kb + f * 16 + ln;
#pragma unroll
      for (int r = 0; r < 8; ++r) {
        float v = s[f][r] * 0.125f;                     // 1/sqrt(64)
        if (key > qbase + r) v = -1e30f;
        s[f][r] = v;
      }
    }
    float sc[8];
#pragma unroll
    for (int r = 0; r < 8; ++r) {
      float bm = fmaxf(s[0][r], s[1][r]);
#pragma unroll
      for (int off = 1; off < 16; off <<= 1) bm = fmaxf(bm, __shfl_xor(bm, off, 32));
      const float mnew = fmaxf(mrow[r], bm);
      sc[r] = __expf(mrow[r] - mnew);
      mrow[r] = mnew;
      const float p0 = __expf(s[0][r] - mnew);
      const float p1 = __expf(s[1][r] - mnew);
      s[0][r] = p0; s[1][r] = p1;
      float rs = p0 + p1;
#pragma unroll
      for (int off = 1; off < 16; off <<= 1) rs += __shfl_xor(rs, off, 32);
      lrow[r] = lrow[r] * sc[r] + rs;
    }
#pragma unroll
    for (int j = 0; j < 4; ++j)
#pragma unroll
      for (int r = 0; r < 8; ++r) o[j][r] *= sc[r];

    // P: C-layout f32 -> LDS (row-major 16x32 bf16) -> A-frag layout
#pragma unroll
    for (int f = 0; f < 2; ++f)
#pragma unroll
      for (int r = 0; r < 8; ++r)
        pt[(r + hi * 8) * 32 + f * 16 + ln] = f2bf(s[f][r]);
    __builtin_amdgcn_wave_barrier();
    const bf16x16 pa = *(const bf16x16*)(pt + ln * 32 + hi * 16);

#pragma unroll
    for (int j = 0; j < 4; ++j) {
      const bf16x16 vf =
          *(const bf16x16*)(vbase + (size_t)(j * 16 + ln) * SEQ + kb + hi * 16);
      o[j] = wmma_bf16(pa, vf, o[j]);
    }
  }

  unsigned short* cb = ctx + ((size_t)b * SEQ + q0) * DM + h * HD;
#pragma unroll
  for (int j = 0; j < 4; ++j)
#pragma unroll
    for (int r = 0; r < 8; ++r)
      cb[(size_t)(r + hi * 8) * DM + j * 16 + ln] = f2bf(o[j][r] / lrow[r]);
}

// ---------------- silu(g) * u -> bf16 ----------------
__global__ __launch_bounds__(256)
void silu_mul(const float* __restrict__ g, const float* __restrict__ u,
              unsigned short* __restrict__ out, size_t n) {
  const size_t i = (size_t)blockIdx.x * blockDim.x + threadIdx.x;
  if (i < n) {
    const float gv = g[i];
    const float s  = gv / (1.f + __expf(-gv));
    out[i] = f2bf(s * u[i]);
  }
}

// ---------------- host launcher ----------------
extern "C" void kernel_launch(void* const* d_in, const int* in_sizes, int n_in,
                              void* d_out, int out_size, void* d_ws, size_t ws_size,
                              hipStream_t stream) {
  const float* x    = (const float*)d_in[0];
  const float* wn1  = (const float*)d_in[1];
  const float* wqkv = (const float*)d_in[2];
  const float* wout = (const float*)d_in[3];
  const float* wn2  = (const float*)d_in[4];
  const float* wg   = (const float*)d_in[5];
  const float* wu   = (const float*)d_in[6];
  const float* wd   = (const float*)d_in[7];
  float* out = (float*)d_out;

  char* ws = (char*)d_ws;
  size_t off = 0;
  auto alloc = [&](size_t bytes) -> void* {
    void* p = ws + off;
    off = (off + bytes + 255) & ~(size_t)255;
    return p;
  };

  unsigned short* wqkvT = (unsigned short*)alloc((size_t)QKVN * DM * 2);
  unsigned short* woutT = (unsigned short*)alloc((size_t)DM * DM * 2);
  unsigned short* wgT   = (unsigned short*)alloc((size_t)DFF * DM * 2);
  unsigned short* wuT   = (unsigned short*)alloc((size_t)DFF * DM * 2);
  unsigned short* wdT   = (unsigned short*)alloc((size_t)DM * DFF * 2);
  unsigned short* hbf   = (unsigned short*)alloc((size_t)MTOK * DM * 2);   // h1 / h2
  float*          qkvf  = (float*)alloc((size_t)MTOK * QKVN * 4);
  unsigned short* qbf   = (unsigned short*)alloc((size_t)BB * NH  * SEQ * HD * 2);
  unsigned short* kbf   = (unsigned short*)alloc((size_t)BB * NKV * SEQ * HD * 2);
  unsigned short* vtbf  = (unsigned short*)alloc((size_t)BB * NKV * HD * SEQ * 2);
  unsigned short* ctxbf = (unsigned short*)alloc((size_t)MTOK * DM * 2);
  float*          xattn = (float*)alloc((size_t)MTOK * DM * 4);
  float*          gf    = (float*)alloc((size_t)MTOK * DFF * 4);
  float*          uf    = (float*)alloc((size_t)MTOK * DFF * 4);
  unsigned short* actbf = (unsigned short*)alloc((size_t)MTOK * DFF * 2);

  const dim3 tb(32, 8);
  transpose_cast<<<dim3(QKVN / 32, DM  / 32), tb, 0, stream>>>(wqkv, wqkvT, DM,  QKVN);
  transpose_cast<<<dim3(DM   / 32, DM  / 32), tb, 0, stream>>>(wout, woutT, DM,  DM);
  transpose_cast<<<dim3(DFF  / 32, DM  / 32), tb, 0, stream>>>(wg,   wgT,   DM,  DFF);
  transpose_cast<<<dim3(DFF  / 32, DM  / 32), tb, 0, stream>>>(wu,   wuT,   DM,  DFF);
  transpose_cast<<<dim3(DM   / 32, DFF / 32), tb, 0, stream>>>(wd,   wdT,   DFF, DM);

  rmsnorm_bf16<<<MTOK, 256, 0, stream>>>(x, wn1, hbf);

  gemm_bf16_wmma<false><<<dim3(QKVN / 128, MTOK / 128), 256, GEMM_LDS_BYTES, stream>>>(
      hbf, wqkvT, nullptr, qkvf, MTOK, QKVN, DM);

  rope_layout<<<BB * SEQ * 48, 64, 0, stream>>>(qkvf, qbf, kbf, vtbf);

  attn_wmma<<<(BB * NH * (SEQ / 16)) / 4, 128, 0, stream>>>(qbf, kbf, vtbf, ctxbf);

  gemm_bf16_wmma<true><<<dim3(DM / 128, MTOK / 128), 256, GEMM_LDS_BYTES, stream>>>(
      ctxbf, woutT, x, xattn, MTOK, DM, DM);

  rmsnorm_bf16<<<MTOK, 256, 0, stream>>>(xattn, wn2, hbf);

  gemm_bf16_wmma<false><<<dim3(DFF / 128, MTOK / 128), 256, GEMM_LDS_BYTES, stream>>>(
      hbf, wgT, nullptr, gf, MTOK, DFF, DM);
  gemm_bf16_wmma<false><<<dim3(DFF / 128, MTOK / 128), 256, GEMM_LDS_BYTES, stream>>>(
      hbf, wuT, nullptr, uf, MTOK, DFF, DM);

  const size_t nact = (size_t)MTOK * DFF;
  silu_mul<<<(unsigned)(nact / 256), 256, 0, stream>>>(gf, uf, actbf, nact);

  gemm_bf16_wmma<true><<<dim3(DM / 128, MTOK / 128), 256, GEMM_LDS_BYTES, stream>>>(
      actbf, wdT, xattn, out, MTOK, DM, DFF);

  (void)in_sizes; (void)n_in; (void)out_size; (void)ws_size;
}
